// VQVAE_52304111731027
// MI455X (gfx1250) — matrix operand order repaired and checked by reading, not compiled, not run on hardware
//
#include <hip/hip_runtime.h>

typedef __attribute__((ext_vector_type(16))) _Float16 v16h;
typedef __attribute__((ext_vector_type(8)))  float    v8f;
typedef __attribute__((ext_vector_type(4)))  unsigned int u32x4;
typedef __attribute__((ext_vector_type(8)))  int      i32x8;
typedef __attribute__((ext_vector_type(4)))  int      i32x4;

#define EPS 1e-5f
#define NBATCH 32

// ---------------------------------------------------------------------------
// Tensor Data Mover: 2-D tile (rows x dim0 elements of f16) -> LDS.
// 6-arg builtin on this toolchain: (u32x4, i32x8, i32x4, i32x4, i32x8, i32).
// ---------------------------------------------------------------------------
__device__ __forceinline__
void tdm_load_2d_f16(const _Float16* gsrc, unsigned ldsOff, int dim0, int rows) {
    unsigned long long ga = (unsigned long long)gsrc;
    u32x4 g0;
    g0[0] = 1u;                                            // count=1, user mode
    g0[1] = ldsOff;                                        // lds_addr (bytes)
    g0[2] = (unsigned)(ga & 0xffffffffull);                // global_addr lo
    g0[3] = (unsigned)((ga >> 32) & 0x01ffffffull) | (2u << 30); // ga hi | type=2
    i32x8 g1;
    g1[0] = (1 << 16);                                     // wg_mask=0, data_size=1 (2B)
    g1[1] = (dim0 & 0xffff) << 16;                         // tensor_dim0[15:0]
    g1[2] = ((dim0 >> 16) & 0xffff) | ((rows & 0xffff) << 16);
    g1[3] = ((rows >> 16) & 0xffff) | ((dim0 & 0xffff) << 16); // | tile_dim0
    g1[4] = (rows & 0xffff);                               // tile_dim1
    g1[5] = dim0;                                          // tensor_dim0_stride
    g1[6] = 0;
    g1[7] = 0;
    i32x4 z4 = {0, 0, 0, 0};
    i32x8 z8 = {0, 0, 0, 0, 0, 0, 0, 0};
    __builtin_amdgcn_tensor_load_to_lds(g0, g1, z4, z4, z8, 0);
}

// ---------------------------------------------------------------------------
// f32 -> f16 convert (network input; C=1 so NCHW == NHWC)
// ---------------------------------------------------------------------------
__global__ __launch_bounds__(256)
void k_f32_to_f16(const float* __restrict__ in, _Float16* __restrict__ out, long n) {
    long i = (long)blockIdx.x * blockDim.x + threadIdx.x;
    if (i < n) out[i] = (_Float16)in[i];
}

// ---------------------------------------------------------------------------
// Pack conv weights into f16 tap-major K matrix [Mpad x Kpad],
// k = (kh*KW + kw)*Ci + ci  (matches NHWC activation runs).
// TRANS=1: source is ConvTranspose2d weight [Ci][Co][KH][KW], flipped.
// ---------------------------------------------------------------------------
template<int Co, int Ci, int KH, int KW, int Kpad, int Mpad, int TRANS>
__global__ __launch_bounds__(256)
void k_wprep(const float* __restrict__ w, _Float16* __restrict__ wmat) {
    constexpr int KHW = KH * KW;
    int i = blockIdx.x * blockDim.x + threadIdx.x;
    if (i >= Mpad * Kpad) return;
    int m = i / Kpad, k = i % Kpad;
    int ci = k % Ci;
    int t  = k / Ci;
    float v = 0.f;
    if (m < Co && t < KHW) {
        int kh = t / KW, kw = t % KW;
        if (!TRANS)
            v = w[((m * Ci + ci) * KH + kh) * KW + kw];
        else
            v = w[((ci * Co + m) * KH + (KH - 1 - kh)) * KW + (KW - 1 - kw)];
    }
    wmat[i] = (_Float16)v;
}

// ---------------------------------------------------------------------------
// Implicit-GEMM conv / conv-transpose, NHWC activations, constant geometry.
// Wave = 16(Co) x 16(pixel); 8 waves/block share a TDM-staged weight tile.
// Ci%16==0: 16-wide B run = one tap, one bounds check, contiguous 32B load.
// ---------------------------------------------------------------------------
template<int Ci, int Hi, int Wi, int Co, int Ho, int Wo,
         int KH, int KW, int STRIDE, int PAD, int TRANS, int Kpad, int OUT_NCHW>
__global__ __launch_bounds__(256)
void k_conv_wmma(const _Float16* __restrict__ act,    // [N,Hi,Wi,Ci] f16
                 const _Float16* __restrict__ wmat,   // [Mpad,Kpad] f16
                 const float* __restrict__ bias,
                 float* __restrict__ out) {           // NHWC (or NCHW if OUT_NCHW)
    constexpr int HoWo = Ho * Wo;
    constexpr int KHW  = KH * KW;

    __shared__ _Float16 wl[16 * Kpad];

    const int lane = threadIdx.x & 31;
    const int wave = threadIdx.x >> 5;
    const int coBase = blockIdx.y * 16;

    if (threadIdx.x == 0) {
        tdm_load_2d_f16(wmat + (size_t)coBase * Kpad,
                        (unsigned)(size_t)&wl[0], Kpad, 16);
        __builtin_amdgcn_s_wait_tensorcnt(0);
    }
    __syncthreads();

    const long pixBase = ((long)blockIdx.x * 8 + wave) * 16;
    const long p  = pixBase + (lane & 15);
    const int img = (int)(p / HoWo);
    const int rem = (int)(p % HoWo);
    const int oh  = rem / Wo, ow = rem % Wo;

    const int mRow = lane & 15;
    const int kA0  = (lane < 16) ? 0 : 16;
    const int kB0  = (lane < 16) ? 0 : 16;

    v8f acc = {};
    for (int k0 = 0; k0 < Kpad; k0 += 32) {
        // ---- A fragment from LDS (2 x b128 per half) ----
        const _Float16* wr = wl + mRow * Kpad + k0;
        v16h a;
        #pragma unroll
        for (int i = 0; i < 8; ++i) a[i]     = wr[kA0 + i];
        #pragma unroll
        for (int i = 0; i < 8; ++i) a[8 + i] = wr[kA0 + 8 + i];

        // ---- B fragment ----
        v16h b;
        if constexpr (Ci % 16 == 0) {
            // 16 consecutive k = one kernel tap, Ci-contiguous in NHWC.
            int kg  = k0 + kB0;
            int t   = kg / Ci;
            int ci0 = kg % Ci;
            int kh = t / KW, kw = t % KW;
            int ih, iw; bool ok;
            if (!TRANS) {
                ih = oh * STRIDE - PAD + kh;
                iw = ow * STRIDE - PAD + kw;
                ok = (ih >= 0) & (ih < Hi) & (iw >= 0) & (iw < Wi);
            } else {
                int th = oh + kh - PAD;
                int tw = ow + kw - PAD;
                ok = (th >= 0) & (tw >= 0) & ((th & 1) == 0) & ((tw & 1) == 0);
                ih = th >> 1; iw = tw >> 1;
                ok = ok & (ih < Hi) & (iw < Wi);
            }
            if (ok) {
                const _Float16* src =
                    act + (((size_t)img * Hi + ih) * Wi + iw) * Ci + ci0;
                #pragma unroll
                for (int i = 0; i < 16; ++i) b[i] = src[i];
            } else {
                #pragma unroll
                for (int i = 0; i < 16; ++i) b[i] = (_Float16)0.f;
            }
        } else {
            // generic per-element gather (only layer 1, Ci==1)
            #pragma unroll
            for (int i = 0; i < 16; ++i) {
                int k = k0 + kB0 + i;
                _Float16 v = (_Float16)0.f;
                int ci = k % Ci;
                int t  = k / Ci;
                if (t < KHW) {
                    int kh = t / KW, kw = t % KW;
                    int ih, iw; bool ok;
                    if (!TRANS) {
                        ih = oh * STRIDE - PAD + kh;
                        iw = ow * STRIDE - PAD + kw;
                        ok = (ih >= 0) & (ih < Hi) & (iw >= 0) & (iw < Wi);
                    } else {
                        int th = oh + kh - PAD;
                        int tw = ow + kw - PAD;
                        ok = (th >= 0) & (tw >= 0) & ((th & 1) == 0) & ((tw & 1) == 0);
                        ih = th >> 1; iw = tw >> 1;
                        ok = ok & (ih < Hi) & (iw < Wi);
                    }
                    if (ok) v = act[(((size_t)img * Hi + ih) * Wi + iw) * Ci + ci];
                }
                b[i] = v;
            }
        }
        acc = __builtin_amdgcn_wmma_f32_16x16x32_f16(false, a, false, b,
                                                     (short)0, acc, false, false);
    }
    // ---- epilogue ----
    const int mOff = (lane < 16) ? 0 : 8;
    if constexpr (OUT_NCHW) {
        #pragma unroll
        for (int r = 0; r < 8; ++r) {
            int co = coBase + mOff + r;
            if (co < Co)
                out[((size_t)img * Co + co) * HoWo + rem] = acc[r] + bias[co];
        }
    } else {
        float* o = out + (size_t)p * Co + coBase + mOff;
        const float* bi = bias + coBase + mOff;
        #pragma unroll
        for (int r = 0; r < 8; ++r) o[r] = acc[r] + bi[r];
    }
}

// ---------------------------------------------------------------------------
// BatchNorm statistics over NHWC: tid -> (pixel-group, channel), coalesced;
// LDS tree over pixel groups, then f32 atomics.
// ---------------------------------------------------------------------------
template<int C, int HW>
__global__ __launch_bounds__(256)
void k_bn_stats(const float* __restrict__ x, float* __restrict__ stats) {
    constexpr int PG = 256 / C;
    const int c  = threadIdx.x % C;
    const int pg = threadIdx.x / C;
    constexpr long NP = (long)NBATCH * HW;
    float s = 0.f, s2 = 0.f;
    for (long pp = (long)blockIdx.x * PG + pg; pp < NP; pp += (long)gridDim.x * PG) {
        float v = x[pp * C + c];
        s += v; s2 += v * v;
    }
    __shared__ float sh[256], sh2[256];
    sh[threadIdx.x] = s; sh2[threadIdx.x] = s2;
    __syncthreads();
    for (int w = PG / 2; w > 0; w >>= 1) {
        if (pg < w) {
            sh[threadIdx.x]  += sh[threadIdx.x + w * C];
            sh2[threadIdx.x] += sh2[threadIdx.x + w * C];
        }
        __syncthreads();
    }
    if (pg == 0) {
        atomicAdd(&stats[c],     sh[c]);
        atomicAdd(&stats[C + c], sh2[c]);
    }
}

// ---------------------------------------------------------------------------
// BN apply (+ReLU) over NHWC; writes f16 NHWC activation.
// ZE=1: also write f32 NCHW copy (z_e_x slot of d_out).
// ---------------------------------------------------------------------------
template<int C, int HW, int RELU, int ZE>
__global__ __launch_bounds__(256)
void k_bn_apply(const float* __restrict__ x, const float* __restrict__ stats,
                const float* __restrict__ g, const float* __restrict__ be,
                _Float16* __restrict__ actOut, float* __restrict__ f32Out) {
    constexpr long total = (long)NBATCH * C * HW;
    long i = (long)blockIdx.x * blockDim.x + threadIdx.x;
    if (i >= total) return;
    int c = (int)(i % C);
    constexpr float cnt = (float)((long)NBATCH * HW);
    float m   = stats[c] / cnt;
    float var = stats[C + c] / cnt - m * m;
    float y = (x[i] - m) * __frsqrt_rn(var + EPS) * g[c] + be[c];
    if (RELU) y = fmaxf(y, 0.f);
    actOut[i] = (_Float16)y;
    if (ZE) {
        long pp = i / C;
        int img = (int)(pp / HW), rem = (int)(pp % HW);
        f32Out[((size_t)img * C + c) * HW + rem] = y;
    }
}

// ---------------------------------------------------------------------------
// VQ helper: negBias[k] = 0.5*||emb_k||^2
// ---------------------------------------------------------------------------
__global__ __launch_bounds__(256)
void k_emb_bias(const float* __restrict__ emb, float* __restrict__ negBias) {
    int k = blockIdx.x * blockDim.x + threadIdx.x;
    if (k >= 512) return;
    float s = 0.f;
    #pragma unroll
    for (int c = 0; c < 32; ++c) { float v = emb[k * 32 + c]; s += v * v; }
    negBias[k] = 0.5f * s;
}

// ---------------------------------------------------------------------------
// VQ argmin: codebook (512x32 f16 = 32KB) TDM-staged in LDS; running WMMA
// argmax; lane-pair combine. zeAct is NHWC so B loads are contiguous.
// ---------------------------------------------------------------------------
__global__ __launch_bounds__(256)
void k_vq_argmin(const _Float16* __restrict__ zeAct,   // [32768,32] NHWC f16
                 const _Float16* __restrict__ embMat,
                 const float* __restrict__ negBias,
                 int* __restrict__ latents, float* __restrict__ latOut) {
    __shared__ _Float16 el[512 * 32];
    if (threadIdx.x == 0) {
        tdm_load_2d_f16(embMat, (unsigned)(size_t)&el[0], 32, 512);
        __builtin_amdgcn_s_wait_tensorcnt(0);
    }
    __syncthreads();

    const int lane = threadIdx.x & 31;
    const int wave = threadIdx.x >> 5;
    const long pixBase = ((long)blockIdx.x * 8 + wave) * 16;
    const long p  = pixBase + (lane & 15);

    const int mRow = lane & 15;
    const int kA0  = (lane < 16) ? 0 : 16;
    const int kB0  = (lane < 16) ? 0 : 16;

    v16h b;
    {
        const _Float16* src = zeAct + (size_t)p * 32 + kB0;
        #pragma unroll
        for (int i = 0; i < 16; ++i) b[i] = src[i];
    }

    float bv = -3.4e38f; int bi = 0;
    const int mOff = (lane < 16) ? 0 : 8;
    for (int m0 = 0; m0 < 512; m0 += 16) {
        const _Float16* er = el + (m0 + mRow) * 32;
        v16h a;
        #pragma unroll
        for (int i = 0; i < 8; ++i) a[i]     = er[kA0 + i];
        #pragma unroll
        for (int i = 0; i < 8; ++i) a[8 + i] = er[kA0 + 8 + i];
        v8f acc = {};
        acc = __builtin_amdgcn_wmma_f32_16x16x32_f16(false, a, false, b,
                                                     (short)0, acc, false, false);
        #pragma unroll
        for (int r = 0; r < 8; ++r) {
            int k = m0 + mOff + r;
            float s = acc[r] - negBias[k];
            if (s > bv) { bv = s; bi = k; }
        }
    }
    float ov = __shfl(bv, (lane & 15) + 16, 32);
    int   oi = __shfl(bi, (lane & 15) + 16, 32);
    if (lane < 16) {
        if (ov > bv) { bv = ov; bi = oi; }
        latents[p] = bi;
        latOut[p]  = (float)bi;
    }
}

// ---------------------------------------------------------------------------
// z_q gather: emb[latent[p]] -> f32 NCHW (d_out) + f16 NHWC (decoder act)
// ---------------------------------------------------------------------------
__global__ __launch_bounds__(256)
void k_zq_gather(const int* __restrict__ latents, const float* __restrict__ emb,
                 float* __restrict__ zqOut, _Float16* __restrict__ zqAct) {
    long i = (long)blockIdx.x * blockDim.x + threadIdx.x;
    if (i >= 32768L * 32) return;
    int p = (int)(i >> 5), c = (int)(i & 31);
    int idx = latents[p];
    float v = emb[idx * 32 + c];
    zqAct[(size_t)p * 32 + c] = (_Float16)v;
    int img = p >> 10, rem = p & 1023;
    zqOut[((size_t)img * 32 + c) * 1024 + rem] = v;
}

// ===========================================================================
// Host launcher
// ===========================================================================
extern "C" void kernel_launch(void* const* d_in, const int* in_sizes, int n_in,
                              void* d_out, int out_size, void* d_ws, size_t ws_size,
                              hipStream_t stream) {
    (void)in_sizes; (void)n_in; (void)out_size; (void)ws_size;
    const float* x   = (const float*)d_in[0];
    const float* w1  = (const float*)d_in[1];  const float* b1  = (const float*)d_in[2];
    const float* g1  = (const float*)d_in[3];  const float* be1 = (const float*)d_in[4];
    const float* w2  = (const float*)d_in[5];  const float* b2  = (const float*)d_in[6];
    const float* g2  = (const float*)d_in[7];  const float* be2 = (const float*)d_in[8];
    const float* w3  = (const float*)d_in[9];  const float* b3  = (const float*)d_in[10];
    const float* g3  = (const float*)d_in[11]; const float* be3 = (const float*)d_in[12];
    const float* w4  = (const float*)d_in[13]; const float* b4  = (const float*)d_in[14];
    const float* g4  = (const float*)d_in[15]; const float* be4 = (const float*)d_in[16];
    const float* emb = (const float*)d_in[17];
    const float* dw1 = (const float*)d_in[18]; const float* db1 = (const float*)d_in[19];
    const float* dg1 = (const float*)d_in[20]; const float* dbe1= (const float*)d_in[21];
    const float* tw1 = (const float*)d_in[22]; const float* tb1 = (const float*)d_in[23];
    const float* tg1 = (const float*)d_in[24]; const float* tbe1= (const float*)d_in[25];
    const float* tw2 = (const float*)d_in[26]; const float* tb2 = (const float*)d_in[27];
    const float* tg2 = (const float*)d_in[28]; const float* tbe2= (const float*)d_in[29];
    const float* tw3 = (const float*)d_in[30]; const float* tb3 = (const float*)d_in[31];

    float* out = (float*)d_out;
    const size_t OFF_ZE = 32ull * 30 * 256 * 256;
    const size_t OFF_ZQ = OFF_ZE + 32ull * 32 * 32 * 32;
    const size_t OFF_LAT= OFF_ZQ + 32ull * 32 * 32 * 32;
    float* xt  = out;
    float* zeF = out + OFF_ZE;
    float* zqF = out + OFF_ZQ;
    float* latF= out + OFF_LAT;

    char* ws = (char*)d_ws;
    size_t off = 0;
    auto carve = [&](size_t bytes) { char* p = ws + off; off = (off + bytes + 255) & ~(size_t)255; return p; };
    _Float16* act0 = (_Float16*)carve(2097152ull * 2);   // x f16 (C=1)
    _Float16* act1 = (_Float16*)carve(8388608ull * 2);   // NHWC from here on
    _Float16* act2 = (_Float16*)carve(4194304ull * 2);
    _Float16* act3 = (_Float16*)carve(2097152ull * 2);
    _Float16* zeA  = (_Float16*)carve(1048576ull * 2);
    _Float16* zqA  = (_Float16*)carve(1048576ull * 2);
    _Float16* d1A  = (_Float16*)carve(2097152ull * 2);
    _Float16* t1A  = (_Float16*)carve(4194304ull * 2);
    _Float16* t2A  = (_Float16*)carve(8388608ull * 2);
    float*    raw  = (float*)   carve(8388608ull * 4);
    _Float16* wmat = (_Float16*)carve(32768ull * 2);
    float*    stats= (float*)   carve(128ull * 4);
    float*    nbias= (float*)   carve(512ull * 4);
    int*      latI = (int*)     carve(32768ull * 4);

    auto wblk = [](long n) { return (unsigned)((n + 255) / 256); };

    // ---- L1: conv(1->16, k4 s2 p1) + BN + ReLU ----
    k_f32_to_f16<<<wblk(2097152), 256, 0, stream>>>(x, act0, 2097152);
    k_wprep<16,1,4,4,32,16,0><<<wblk(16*32), 256, 0, stream>>>(w1, wmat);
    k_conv_wmma<1,256,256, 16,128,128, 4,4,2,1,0, 32, 0>
        <<<dim3(4096,1), 256, 0, stream>>>(act0, wmat, b1, raw);
    (void)hipMemsetAsync(stats, 0, 128 * sizeof(float), stream);
    k_bn_stats<16,16384><<<64, 256, 0, stream>>>(raw, stats);
    k_bn_apply<16,16384,1,0><<<wblk(8388608), 256, 0, stream>>>(raw, stats, g1, be1, act1, nullptr);

    // ---- L2: conv(16->32) ----
    k_wprep<32,16,4,4,256,32,0><<<wblk(32*256), 256, 0, stream>>>(w2, wmat);
    k_conv_wmma<16,128,128, 32,64,64, 4,4,2,1,0, 256, 0>
        <<<dim3(1024,2), 256, 0, stream>>>(act1, wmat, b2, raw);
    (void)hipMemsetAsync(stats, 0, 128 * sizeof(float), stream);
    k_bn_stats<32,4096><<<64, 256, 0, stream>>>(raw, stats);
    k_bn_apply<32,4096,1,0><<<wblk(4194304), 256, 0, stream>>>(raw, stats, g2, be2, act2, nullptr);

    // ---- L3: conv(32->64) ----
    k_wprep<64,32,4,4,512,64,0><<<wblk(64*512), 256, 0, stream>>>(w3, wmat);
    k_conv_wmma<32,64,64, 64,32,32, 4,4,2,1,0, 512, 0>
        <<<dim3(256,4), 256, 0, stream>>>(act2, wmat, b3, raw);
    (void)hipMemsetAsync(stats, 0, 128 * sizeof(float), stream);
    k_bn_stats<64,1024><<<64, 256, 0, stream>>>(raw, stats);
    k_bn_apply<64,1024,1,0><<<wblk(2097152), 256, 0, stream>>>(raw, stats, g3, be3, act3, nullptr);

    // ---- L4: 1x1 conv(64->32) + BN -> z_e_x (f16 NHWC + f32 NCHW) ----
    k_wprep<32,64,1,1,64,32,0><<<wblk(32*64), 256, 0, stream>>>(w4, wmat);
    k_conv_wmma<64,32,32, 32,32,32, 1,1,1,0,0, 64, 0>
        <<<dim3(256,2), 256, 0, stream>>>(act3, wmat, b4, raw);
    (void)hipMemsetAsync(stats, 0, 128 * sizeof(float), stream);
    k_bn_stats<32,1024><<<64, 256, 0, stream>>>(raw, stats);
    k_bn_apply<32,1024,0,1><<<wblk(1048576), 256, 0, stream>>>(raw, stats, g4, be4, zeA, zeF);

    // ---- VQ ----
    k_wprep<512,32,1,1,32,512,0><<<wblk(512*32), 256, 0, stream>>>(emb, wmat);
    k_emb_bias<<<wblk(512), 256, 0, stream>>>(emb, nbias);
    k_vq_argmin<<<dim3(256), 256, 0, stream>>>(zeA, wmat, nbias, latI, latF);
    k_zq_gather<<<wblk(1048576), 256, 0, stream>>>(latI, emb, zqF, zqA);

    // ---- D1: 1x1 conv(32->64) + BN + ReLU ----
    k_wprep<64,32,1,1,32,64,0><<<wblk(64*32), 256, 0, stream>>>(dw1, wmat);
    k_conv_wmma<32,32,32, 64,32,32, 1,1,1,0,0, 32, 0>
        <<<dim3(256,4), 256, 0, stream>>>(zqA, wmat, db1, raw);
    (void)hipMemsetAsync(stats, 0, 128 * sizeof(float), stream);
    k_bn_stats<64,1024><<<64, 256, 0, stream>>>(raw, stats);
    k_bn_apply<64,1024,1,0><<<wblk(2097152), 256, 0, stream>>>(raw, stats, dg1, dbe1, d1A, nullptr);

    // ---- T1: convT(64->32, k4 s2 p1) ----
    k_wprep<32,64,4,4,1024,32,1><<<wblk(32*1024), 256, 0, stream>>>(tw1, wmat);
    k_conv_wmma<64,32,32, 32,64,64, 4,4,2,2,1, 1024, 0>
        <<<dim3(1024,2), 256, 0, stream>>>(d1A, wmat, tb1, raw);
    (void)hipMemsetAsync(stats, 0, 128 * sizeof(float), stream);
    k_bn_stats<32,4096><<<64, 256, 0, stream>>>(raw, stats);
    k_bn_apply<32,4096,1,0><<<wblk(4194304), 256, 0, stream>>>(raw, stats, tg1, tbe1, t1A, nullptr);

    // ---- T2: convT(32->16) ----
    k_wprep<16,32,4,4,512,16,1><<<wblk(16*512), 256, 0, stream>>>(tw2, wmat);
    k_conv_wmma<32,64,64, 16,128,128, 4,4,2,2,1, 512, 0>
        <<<dim3(4096,1), 256, 0, stream>>>(t1A, wmat, tb2, raw);
    (void)hipMemsetAsync(stats, 0, 128 * sizeof(float), stream);
    k_bn_stats<16,16384><<<64, 256, 0, stream>>>(raw, stats);
    k_bn_apply<16,16384,1,0><<<wblk(8388608), 256, 0, stream>>>(raw, stats, tg2, tbe2, t2A, nullptr);

    // ---- T3: convT(16->30) -> x_tilde (NCHW) directly ----
    k_wprep<30,16,4,4,256,32,1><<<wblk(32*256), 256, 0, stream>>>(tw3, wmat);
    k_conv_wmma<16,128,128, 30,256,256, 4,4,2,2,1, 256, 1>
        <<<dim3(16384,2), 256, 0, stream>>>(t2A, wmat, tb3, xt);
}
